// NetSIR_67053029425214
// MI455X (gfx1250) — matrix operand order repaired and checked by reading, not compiled, not run on hardware
//
#include <hip/hip_runtime.h>
#include <stdint.h>

// Network SIR: 199 sequential steps of y = adj @ I (5000x5000 dense) + elementwise update.
// MI455X reasoning:
//  - adj converted once to bf16 (~51 MB) -> resident in 192 MB global L2; each step
//    streams from L2, not HBM (kernel is ~0.5 FLOP/byte, pure bandwidth -> halve bytes).
//  - adj is stored PRE-SWIZZLED in exact v_wmma_f32_16x16x32_bf16 A-operand lane order:
//    each 16x32 tile is 1 KB contiguous, so a wave loads two contiguous 512 B bursts.
//  - K padded to 5120 so the 2x-unrolled loop (two independent WMMA accumulators for ILP)
//    runs exactly 20 uniform iterations per wave, no guards.
//  - B operand = I-chunk replicated across all 16 columns -> every D column equals y,
//    EXEC stays all-ones (WMMA requirement), lanes 0/16 hold the 16 row sums.
//  - Sequential step dependency -> per-step kernels on `stream` (graph-capturable).

#define N_NODES 5000
#define K_PAD   5120   // multiple of 256 = 4 waves * 2 chunks * 32 (uniform unrolled loop)
#define R_PAD   5008   // multiple of 16 (WMMA M)
#define TILE_E  512    // elements per 16x32 bf16 tile (1 KB)

typedef __attribute__((ext_vector_type(16))) __bf16 v16bf;
typedef __attribute__((ext_vector_type(8)))  float  v8f;

static __device__ __forceinline__ uint16_t f32_to_bf16(float f) {
    uint32_t u = __float_as_uint(f);
    uint32_t r = u + 0x7FFFu + ((u >> 16) & 1u);   // round-to-nearest-even
    return (uint16_t)(r >> 16);
}

// One-time: fp32 adj -> bf16, swizzled into WMMA A-operand tile order.
// Tile (ib, jc) = rows 16*ib.., K 32*jc.. ; within a tile, element p:
//   half = p>>8, L = (p&255)>>3, e = p&7  ->  lane L, load {0,1}, element e
//   m = L&15 ; k = (L<16 ? 0 : 8) + half*16 + e    (ISA 7.12.2 16-bit A layout)
__global__ void convert_adj_kernel(const float* __restrict__ adj,
                                   uint16_t* __restrict__ out) {
    size_t idx = (size_t)blockIdx.x * blockDim.x + threadIdx.x;
    const size_t total = (size_t)R_PAD * K_PAD;
    if (idx >= total) return;
    const size_t rowblk_sz = (size_t)K_PAD * 16;
    int ib = (int)(idx / rowblk_sz);
    int rem = (int)(idx % rowblk_sz);
    int jc = rem / TILE_E;
    int p  = rem % TILE_E;
    int half = p >> 8;
    int q = p & 255;
    int L = q >> 3;
    int e = q & 7;
    int m = L & 15;
    int k = ((L < 16) ? 0 : 8) + (half << 4) + e;
    int r = ib * 16 + m;
    int c = jc * 32 + k;
    uint16_t v = 0;
    if (r < N_NODES && c < N_NODES) v = f32_to_bf16(adj[(size_t)r * N_NODES + c]);
    out[idx] = v;
}

// Trajectory step 0 = x.
__global__ void copy_x_kernel(const float* __restrict__ x, float* __restrict__ out) {
    int i = blockIdx.x * blockDim.x + threadIdx.x;
    if (i < N_NODES * 3) out[i] = x[i];
}

// Extract I column of current state (stride-3) into dense, zero-padded bf16 vector.
__global__ void extract_I_kernel(const float* __restrict__ state,
                                 uint16_t* __restrict__ Ibf) {
    int n = blockIdx.x * blockDim.x + threadIdx.x;
    if (n >= K_PAD) return;
    float v = (n < N_NODES) ? state[n * 3 + 1] : 0.0f;
    Ibf[n] = f32_to_bf16(v);
}

// One block = 16 output rows; 4 waves split K (2 chunks each per iteration, 2 independent
// accumulators); LDS reduction; 16 lead threads apply the SIR update.
__global__ __launch_bounds__(128)
void sir_step_kernel(const uint16_t* __restrict__ adjs,   // swizzled [R_PAD/16][K_PAD/32][512]
                     const uint16_t* __restrict__ Ibf,    // [K_PAD] bf16
                     const float* __restrict__ prev,      // state t   (N,3)
                     float* __restrict__ next,            // state t+1 (N,3)
                     const float* __restrict__ beta,
                     const float* __restrict__ gamma) {
    __shared__ float red[4][16];

    const int tid  = threadIdx.x;
    const int lane = tid & 31;
    const int wave = tid >> 5;
    const int r0   = blockIdx.x * 16;

    // Swizzled A: wave's first chunk pair starts at K = wave*64; tile jc lives at
    // element offset jc*512 = k*16. Per-lane slice is 16B-aligned contiguous.
    const uint16_t* pA = adjs + (size_t)blockIdx.x * ((size_t)K_PAD * 16)
                              + (size_t)wave * 64 * 16 + lane * 8;
    // B layout (16-bit 32x16): lanes 0-15 hold K=0..7,16..23, lanes 16-31 hold K=8..15,24..31.
    const uint16_t* pB = Ibf + wave * 64 + ((lane < 16) ? 0 : 8);

    v8f acc0 = {};
    v8f acc1 = {};
    #pragma unroll 4
    for (int it = 0; it < K_PAD / 256; ++it) {
        union { uint4 u[2]; v16bf v; } A0, A1, B0, B1;
        A0.u[0] = *(const uint4*)(pA +   0);
        A0.u[1] = *(const uint4*)(pA + 256);
        A1.u[0] = *(const uint4*)(pA + 512);
        A1.u[1] = *(const uint4*)(pA + 768);
        B0.u[0] = *(const uint4*)(pB +  0);
        B0.u[1] = *(const uint4*)(pB + 16);
        B1.u[0] = *(const uint4*)(pB + 32);
        B1.u[1] = *(const uint4*)(pB + 48);
        acc0 = __builtin_amdgcn_wmma_f32_16x16x32_bf16(false, A0.v, false, B0.v,
                                                       (short)0, acc0, false, false);
        acc1 = __builtin_amdgcn_wmma_f32_16x16x32_bf16(false, A1.v, false, B1.v,
                                                       (short)0, acc1, false, false);
        pA += 4096;   // 4 waves * 2 tiles * 512 elements
        pB += 256;
    }
    v8f acc = acc0 + acc1;

    // D layout: lane 0 VGPR r = (M=r, N=0); lane 16 VGPR r = (M=r+8, N=0).
    if (lane == 0 || lane == 16) {
        const int half = (lane >> 4) * 8;
        #pragma unroll
        for (int j = 0; j < 8; ++j) red[wave][half + j] = acc[j];
    }
    __syncthreads();

    if (tid < 16) {
        const float y = red[0][tid] + red[1][tid] + red[2][tid] + red[3][tid];
        const int r = r0 + tid;
        if (r < N_NODES) {
            const float S = prev[r * 3 + 0];
            const float I = prev[r * 3 + 1];
            const float R = prev[r * 3 + 2];
            const float nc = beta[r] * (S * y);
            const float nr = gamma[r] * I;
            next[r * 3 + 0] = S - nc;
            next[r * 3 + 1] = I + nc - nr;
            next[r * 3 + 2] = R + nr;
        }
    }
}

extern "C" void kernel_launch(void* const* d_in, const int* in_sizes, int n_in,
                              void* d_out, int out_size, void* d_ws, size_t ws_size,
                              hipStream_t stream) {
    const float* x     = (const float*)d_in[0];
    const float* adj   = (const float*)d_in[1];
    const float* beta  = (const float*)d_in[2];
    const float* gamma = (const float*)d_in[3];
    float* out = (float*)d_out;

    const int steps = out_size / (N_NODES * 3);   // 200, derived deterministically

    // Workspace layout: [swizzled bf16 adj | bf16 I vector]
    uint16_t* adjs = (uint16_t*)d_ws;
    const size_t adj_bytes = (size_t)R_PAD * K_PAD * sizeof(uint16_t);  // ~51.3 MB
    uint16_t* Ibf = (uint16_t*)((char*)d_ws + ((adj_bytes + 255) & ~(size_t)255));

    const size_t conv_total = (size_t)R_PAD * K_PAD;
    convert_adj_kernel<<<(unsigned)((conv_total + 255) / 256), 256, 0, stream>>>(adj, adjs);
    copy_x_kernel<<<(N_NODES * 3 + 255) / 256, 256, 0, stream>>>(x, out);

    for (int t = 0; t < steps - 1; ++t) {
        const float* prev = out + (size_t)t * N_NODES * 3;
        float*       nxt  = out + (size_t)(t + 1) * N_NODES * 3;
        extract_I_kernel<<<(K_PAD + 255) / 256, 256, 0, stream>>>(prev, Ibf);
        sir_step_kernel<<<(N_NODES + 15) / 16, 128, 0, stream>>>(adjs, Ibf, prev, nxt,
                                                                 beta, gamma);
    }
}
